// AttentionBlock_37263136260339
// MI455X (gfx1250) — compile-verified, matrix operand order checked by reading, not run on hardware
//
#include <hip/hip_runtime.h>
#include <math.h>

// ---------------- problem constants ----------------
#define BB    16
#define IMGSZ 224
#define HO    112
#define WOUT  112
#define NTOK  (HO*WOUT)       // 12544
#define CDIM  128
#define HIDD  512
#define NT    (BB*NTOK)       // 200704 tokens total

typedef __attribute__((ext_vector_type(16))) _Float16 v16h;
typedef __attribute__((ext_vector_type(8)))  _Float16 v8h;
typedef __attribute__((ext_vector_type(8)))  float    v8f;
typedef __attribute__((ext_vector_type(4)))  unsigned v4u;
typedef __attribute__((ext_vector_type(8)))  unsigned v8u;

// Build a 16-wide f16 fragment from two contiguous 8-half (16B) loads.
__device__ __forceinline__ v16h frag_cat(v8h lo, v8h hi) {
  v16h r;
#pragma unroll
  for (int i = 0; i < 8; ++i) { r[i] = lo[i]; r[i + 8] = hi[i]; }
  return r;
}

// Row-contiguous fragment load for the CDNA5 16-bit WMMA operand layout.
// lane<16: elements 0..7 = K{kb..kb+7}, 8..15 = K{kb+16..kb+23}; kb = k0 (+8 for lanes>=16).
__device__ __forceinline__ v16h load_frag(const _Float16* __restrict__ p, int ld,
                                          int row, int kb) {
  const _Float16* q = p + (size_t)row * ld + kb;
  v8h lo = *(const v8h*)(q);
  v8h hi = *(const v8h*)(q + 16);
  return frag_cat(lo, hi);
}

// 16x64 per-wave tile: one A fragment feeds 4 WMMAs (4 N-tiles).
template <int KTOT>
__device__ __forceinline__ void gemm_1x4(const _Float16* __restrict__ A, int lda, int mrow,
                                         const _Float16* __restrict__ WT, int ldb,
                                         int nrowBase, int l15, int hf, v8f acc[4]) {
#pragma unroll
  for (int k0 = 0; k0 < KTOT; k0 += 32) {
    int kb = k0 + hf * 8;
    v16h a = load_frag(A, lda, mrow, kb);
#pragma unroll
    for (int j = 0; j < 4; ++j) {
      v16h bf = load_frag(WT, ldb, nrowBase + j * 16 + l15, kb);
      acc[j] = __builtin_amdgcn_wmma_f32_16x16x32_f16(false, a, false, bf, (short)0,
                                                      acc[j], false, false);
    }
  }
}

// ---- Tensor Data Mover: 2D tile load (32 rows x 128 f16, row stride 128),
// padded in LDS by 4 DWORDs every 64 DWORDs -> matches [32][136] layout.
__device__ __forceinline__ void tdm_load_tile_32x128(unsigned ldsOff, const _Float16* gsrc) {
  unsigned long long ga = (unsigned long long)(size_t)gsrc;
  v4u g0 = { 1u,                                   // count=1, user descriptor
             ldsOff,                               // lds_addr (bytes)
             (unsigned)ga,                         // global_addr[31:0]
             (unsigned)(ga >> 32) | 0x80000000u }; // global_addr[56:32] | type=2
  v8u g1 = { (1u << 16)                            // data_size = 2 bytes
               | (1u << 20)                        // pad_enable
               | (5u << 22)                        // pad_interval: 64 DWORDs
               | (3u << 25),                       // pad_amount: 4 DWORDs
             128u << 16,                           // tensor_dim0 = 128
             ((unsigned)NTOK & 0xffffu) << 16,     // tensor_dim1.lo = 12544
             128u << 16,                           // tensor_dim1.hi=0 | tile_dim0=128
             32u,                                  // tile_dim1 = 32, tile_dim2 = 0
             128u,                                 // tensor_dim0_stride = 128
             0u, 0u };
  asm volatile("tensor_load_to_lds %0, %1" :: "s"(g0), "s"(g1) : "memory");
}

// ---------------- weight transpose f32 -> f16 (dst[n*K+k] = src[k*N+n]) ---
__global__ void wtrans_kernel(const float* __restrict__ src, _Float16* __restrict__ dst,
                              int K, int Nc) {
  int idx = blockIdx.x * blockDim.x + threadIdx.x;
  if (idx >= K * Nc) return;
  int n = idx % Nc, k = idx / Nc;
  dst[(size_t)n * K + k] = (_Float16)src[(size_t)k * Nc + n];
}

// ---------------- patch embed: 7x7 stride2 pad3, Cin=1 -> 128 -------------
__global__ __launch_bounds__(256)
void patch_embed_kernel(const float* __restrict__ x, const float* __restrict__ pew,
                        const float* __restrict__ peb, float* __restrict__ y) {
  __shared__ float xs[7][IMGSZ + 8];
  __shared__ float ws[CDIM * 49];
  int b  = blockIdx.x / HO;
  int oy = blockIdx.x % HO;
  for (int i = threadIdx.x; i < CDIM * 49; i += 256) ws[i] = pew[i];
  for (int i = threadIdx.x; i < 7 * IMGSZ; i += 256) {
    int r = i / IMGSZ, cidx = i % IMGSZ;
    int iy = 2 * oy - 3 + r;
    xs[r][cidx] = (iy >= 0 && iy < IMGSZ) ? x[((size_t)b * IMGSZ + iy) * IMGSZ + cidx] : 0.f;
  }
  __syncthreads();
  for (int i = threadIdx.x; i < WOUT * CDIM; i += 256) {
    int ox = i % WOUT;
    int oc = i / WOUT;
    float acc = peb[oc];
    const float* wp = &ws[oc * 49];
#pragma unroll
    for (int ky = 0; ky < 7; ++ky) {
#pragma unroll
      for (int kx = 0; kx < 7; ++kx) {
        int ix = 2 * ox - 3 + kx;
        float xv = (ix >= 0 && ix < IMGSZ) ? xs[ky][ix] : 0.f;
        acc += xv * wp[ky * 7 + kx];
      }
    }
    y[(((size_t)b * CDIM + oc) * HO + oy) * WOUT + ox] = acc;
  }
}

// -------- depthwise 3x3 gate: t[b,n,c] = y * sigmoid(dw(y)) (scatter) -----
__global__ void posgate_kernel(const float* __restrict__ y, const float* __restrict__ pw,
                               const float* __restrict__ pb, float* __restrict__ t) {
  size_t idx = (size_t)blockIdx.x * blockDim.x + threadIdx.x;
  if (idx >= (size_t)BB * CDIM * NTOK) return;
  int xq = (int)(idx % WOUT);
  int yq = (int)((idx / WOUT) % HO);
  int c  = (int)((idx / NTOK) % CDIM);
  int b  = (int)(idx / ((size_t)NTOK * CDIM));
  const float* yp = y + ((size_t)b * CDIM + c) * NTOK;
  float acc = pb[c];
#pragma unroll
  for (int ky = 0; ky < 3; ++ky) {
    int yy = yq + ky - 1; if (yy < 0 || yy >= HO) continue;
#pragma unroll
    for (int kx = 0; kx < 3; ++kx) {
      int xx = xq + kx - 1; if (xx < 0 || xx >= WOUT) continue;
      acc += yp[yy * WOUT + xx] * pw[c * 9 + ky * 3 + kx];
    }
  }
  float center = yp[yq * WOUT + xq];
  float g = center * (1.0f / (1.0f + __expf(-acc)));
  t[((size_t)b * NTOK + (size_t)yq * WOUT + xq) * CDIM + c] = g;
}

// ------ per-token: t = LN(t, pe_ln); h16 = f16(LN(t, ln1)) ----------------
__global__ __launch_bounds__(128)
void tokln_kernel(float* __restrict__ t, const float* __restrict__ g0,
                  const float* __restrict__ b0, const float* __restrict__ g1,
                  const float* __restrict__ b1, _Float16* __restrict__ h16) {
  __shared__ float red[128];
  size_t tok = blockIdx.x;
  int c = threadIdx.x;
  float v = t[tok * CDIM + c];
  red[c] = v; __syncthreads();
  for (int s = 64; s > 0; s >>= 1) { if (c < s) red[c] += red[c + s]; __syncthreads(); }
  float mu = red[0] * (1.0f / CDIM); __syncthreads();
  float d = v - mu;
  red[c] = d * d; __syncthreads();
  for (int s = 64; s > 0; s >>= 1) { if (c < s) red[c] += red[c + s]; __syncthreads(); }
  float var = red[0] * (1.0f / CDIM); __syncthreads();
  float tn = d * rsqrtf(var + 1e-5f) * g0[c] + b0[c];
  t[tok * CDIM + c] = tn;
  red[c] = tn; __syncthreads();
  for (int s = 64; s > 0; s >>= 1) { if (c < s) red[c] += red[c + s]; __syncthreads(); }
  float mu2 = red[0] * (1.0f / CDIM); __syncthreads();
  float d2 = tn - mu2;
  red[c] = d2 * d2; __syncthreads();
  for (int s = 64; s > 0; s >>= 1) { if (c < s) red[c] += red[c + s]; __syncthreads(); }
  float var2 = red[0] * (1.0f / CDIM);
  h16[tok * CDIM + c] = (_Float16)(d2 * rsqrtf(var2 + 1e-6f) * g1[c] + b1[c]);
}

// ---- generic single LN -> f16 --------------------------------------------
__global__ __launch_bounds__(128)
void ln_kernel(const float* __restrict__ t, const float* __restrict__ g,
               const float* __restrict__ bt, _Float16* __restrict__ out, float eps) {
  __shared__ float red[128];
  size_t tok = blockIdx.x;
  int c = threadIdx.x;
  float v = t[tok * CDIM + c];
  red[c] = v; __syncthreads();
  for (int s = 64; s > 0; s >>= 1) { if (c < s) red[c] += red[c + s]; __syncthreads(); }
  float mu = red[0] * (1.0f / CDIM); __syncthreads();
  float d = v - mu;
  red[c] = d * d; __syncthreads();
  for (int s = 64; s > 0; s >>= 1) { if (c < s) red[c] += red[c + s]; __syncthreads(); }
  float var = red[0] * (1.0f / CDIM);
  out[tok * CDIM + c] = (_Float16)(d * rsqrtf(var + eps) * g[c] + bt[c]);
}

// ---- fused qkv GEMM: [NT,128] x [128,384], epilogues elu+1 / /N ----------
__global__ __launch_bounds__(256)
void qkv_kernel(const _Float16* __restrict__ h16, const _Float16* __restrict__ wT,
                const float* __restrict__ q_b, const float* __restrict__ kv_b,
                _Float16* __restrict__ Q16, _Float16* __restrict__ K16,
                _Float16* __restrict__ V16) {
  int lane = threadIdx.x & 31, wave = threadIdx.x >> 5;
  int tile = blockIdx.x * 8 + wave;
  const int NQ = 384 / 64;                 // 6 quad-tiles
  int mt = tile / NQ, nq = tile % NQ;
  int l15 = lane & 15, hf = lane >> 4;
  v8f zero = {};
  v8f acc[4] = { zero, zero, zero, zero };
  gemm_1x4<128>(h16, CDIM, mt * 16 + l15, wT, CDIM, nq * 64, l15, hf, acc);
#pragma unroll
  for (int j = 0; j < 4; ++j) {
    int col = nq * 64 + j * 16 + l15;
    float bias = (col < 128) ? q_b[col] : kv_b[col - 128];
#pragma unroll
    for (int i = 0; i < 8; ++i) {
      int m = mt * 16 + hf * 8 + i;
      float v = acc[j][i] + bias;
      if (col < 128) {
        float o = v > 0.f ? v + 1.f : __expf(v);          // elu(v)+1
        Q16[(size_t)m * CDIM + col] = (_Float16)o;
      } else if (col < 256) {
        float o = v > 0.f ? v + 1.f : __expf(v);
        K16[(size_t)m * CDIM + (col - 128)] = (_Float16)o;
      } else {
        V16[(size_t)m * CDIM + (col - 256)] = (_Float16)(v * (1.0f / NTOK));
      }
    }
  }
}

// ---- Ksum[b,c] = sum_s K16[b,s,c] ----------------------------------------
__global__ __launch_bounds__(128)
void ksum_kernel(const _Float16* __restrict__ K16, float* __restrict__ ksum) {
  int b = blockIdx.x, c = threadIdx.x;
  float s = 0.f;
  const _Float16* p = K16 + (size_t)b * NTOK * CDIM + c;
  for (int i = 0; i < NTOK; ++i) s += (float)p[(size_t)i * CDIM];
  ksum[b * CDIM + c] = s;
}

// ---- per-batch KV = K^T V (128x128, K-dim = 12544), TDM-staged WMMA ------
__global__ __launch_bounds__(256)
void kv_outer_kernel(const _Float16* __restrict__ K16, const _Float16* __restrict__ V16,
                     _Float16* __restrict__ KVT16) {
  __shared__ _Float16 Kc[32][CDIM + 8];
  __shared__ _Float16 Vc[32][CDIM + 8];
  int b = blockIdx.x;
  int lane = threadIdx.x & 31, wave = threadIdx.x >> 5;
  int l15 = lane & 15, hf = lane >> 4;
  v8f zero = {};
  v8f acc[8];
#pragma unroll
  for (int i = 0; i < 8; ++i) acc[i] = zero;
  const size_t base = (size_t)b * NTOK * CDIM;
  unsigned kcOff = (unsigned)(size_t)&Kc[0][0];
  unsigned vcOff = (unsigned)(size_t)&Vc[0][0];
  int mBase = wave * 16;
  for (int s0 = 0; s0 < NTOK; s0 += 32) {
    __syncthreads();                       // previous iteration's reads done
    if (wave == 0) {
      tdm_load_tile_32x128(kcOff, K16 + base + (size_t)s0 * CDIM);
      tdm_load_tile_32x128(vcOff, V16 + base + (size_t)s0 * CDIM);
      __builtin_amdgcn_s_wait_tensorcnt(0);
    }
    __syncthreads();
    int kb = hf * 8;
    v16h a;
#pragma unroll
    for (int j = 0; j < 8; ++j) {
      a[j]     = Kc[kb + j][mBase + l15];
      a[j + 8] = Kc[kb + 16 + j][mBase + l15];
    }
#pragma unroll
    for (int nt2 = 0; nt2 < 8; ++nt2) {
      v16h bf;
#pragma unroll
      for (int j = 0; j < 8; ++j) {
        bf[j]     = Vc[kb + j][nt2 * 16 + l15];
        bf[j + 8] = Vc[kb + 16 + j][nt2 * 16 + l15];
      }
      acc[nt2] = __builtin_amdgcn_wmma_f32_16x16x32_f16(false, a, false, bf, (short)0,
                                                        acc[nt2], false, false);
    }
  }
  // store transposed (KVT[n][m]) so the attn GEMM B-fragments are contiguous
  _Float16* out = KVT16 + (size_t)b * CDIM * CDIM;
#pragma unroll
  for (int nt2 = 0; nt2 < 8; ++nt2) {
#pragma unroll
    for (int i = 0; i < 8; ++i) {
      int m = mBase + hf * 8 + i;
      int n = nt2 * 16 + l15;
      out[(size_t)n * CDIM + m] = (_Float16)acc[nt2][i];
    }
  }
}

// ---- Z[tok] = 1/(Q . Ksum + eps) -----------------------------------------
__global__ void zcalc_kernel(const _Float16* __restrict__ Q16, const float* __restrict__ ksum,
                             float* __restrict__ zbuf) {
  int tok = blockIdx.x * blockDim.x + threadIdx.x;
  if (tok >= NT) return;
  int b = tok / NTOK;
  const _Float16* q = Q16 + (size_t)tok * CDIM;
  const float* ks = ksum + b * CDIM;
  float s = 0.f;
  for (int d = 0; d < CDIM; ++d) s += (float)q[d] * ks[d];
  zbuf[tok] = 1.0f / (s + 1e-6f);
}

// ---- attn: t += (Q @ KV) * Z * N -----------------------------------------
__global__ __launch_bounds__(256)
void attn_kernel(const _Float16* __restrict__ Q16, const _Float16* __restrict__ KVT16,
                 const float* __restrict__ zbuf, float* __restrict__ t) {
  int lane = threadIdx.x & 31, wave = threadIdx.x >> 5;
  int tile = blockIdx.x * 8 + wave;
  int mt = tile >> 1, nq = tile & 1;       // 2 quad-tiles cover 128 cols
  int l15 = lane & 15, hf = lane >> 4;
  int b = (mt * 16) / NTOK;
  const _Float16* kvt = KVT16 + (size_t)b * CDIM * CDIM;
  v8f zero = {};
  v8f acc[4] = { zero, zero, zero, zero };
  gemm_1x4<128>(Q16, CDIM, mt * 16 + l15, kvt, CDIM, nq * 64, l15, hf, acc);
#pragma unroll
  for (int j = 0; j < 4; ++j) {
    int col = nq * 64 + j * 16 + l15;
#pragma unroll
    for (int i = 0; i < 8; ++i) {
      int m = mt * 16 + hf * 8 + i;
      float z = zbuf[m] * (float)NTOK;
      t[(size_t)m * CDIM + col] += acc[j][i] * z;
    }
  }
}

// ---- fc1: [NT,128]x[128,512] + bias -> f32 hmid --------------------------
__global__ __launch_bounds__(256)
void fc1_kernel(const _Float16* __restrict__ h16, const _Float16* __restrict__ wT,
                const float* __restrict__ bias, float* __restrict__ hmid) {
  int lane = threadIdx.x & 31, wave = threadIdx.x >> 5;
  int tile = blockIdx.x * 8 + wave;
  const int NQ = HIDD / 64;                // 8 quad-tiles
  int mt = tile / NQ, nq = tile % NQ;
  int l15 = lane & 15, hf = lane >> 4;
  v8f zero = {};
  v8f acc[4] = { zero, zero, zero, zero };
  gemm_1x4<128>(h16, CDIM, mt * 16 + l15, wT, CDIM, nq * 64, l15, hf, acc);
#pragma unroll
  for (int j = 0; j < 4; ++j) {
    int col = nq * 64 + j * 16 + l15;
    float bb = bias[col];
#pragma unroll
    for (int i = 0; i < 8; ++i) {
      int m = mt * 16 + hf * 8 + i;
      hmid[(size_t)m * HIDD + col] = acc[j][i] + bb;
    }
  }
}

// ---- depthwise 3x3 on token-major (B,HW,512) + exact GELU -> f16 ---------
__global__ void dwgelu_kernel(const float* __restrict__ hmid, const float* __restrict__ dww,
                              const float* __restrict__ dwb, _Float16* __restrict__ hact) {
  size_t idx = (size_t)blockIdx.x * blockDim.x + threadIdx.x;
  if (idx >= (size_t)NT * HIDD) return;
  int c = (int)(idx & (HIDD - 1));
  size_t tok = idx >> 9;
  int hw = (int)(tok % NTOK);
  size_t boff = tok - hw;                  // b*NTOK
  int yy0 = hw / WOUT, xx0 = hw % WOUT;
  float accv = dwb[c];
#pragma unroll
  for (int ky = 0; ky < 3; ++ky) {
    int yy = yy0 + ky - 1; if (yy < 0 || yy >= HO) continue;
#pragma unroll
    for (int kx = 0; kx < 3; ++kx) {
      int xx = xx0 + kx - 1; if (xx < 0 || xx >= WOUT) continue;
      accv += hmid[(boff + (size_t)yy * WOUT + xx) * HIDD + c] * dww[c * 9 + ky * 3 + kx];
    }
  }
  float g = 0.5f * accv * (1.0f + erff(accv * 0.70710678f));
  hact[idx] = (_Float16)g;
}

// ---- fc2: [NT,512]x[512,128] + bias, residual add into t -----------------
__global__ __launch_bounds__(256)
void fc2_kernel(const _Float16* __restrict__ hact, const _Float16* __restrict__ wT,
                const float* __restrict__ bias, float* __restrict__ t) {
  int lane = threadIdx.x & 31, wave = threadIdx.x >> 5;
  int tile = blockIdx.x * 8 + wave;
  int mt = tile >> 1, nq = tile & 1;       // 2 quad-tiles cover 128 cols
  int l15 = lane & 15, hf = lane >> 4;
  v8f zero = {};
  v8f acc[4] = { zero, zero, zero, zero };
  gemm_1x4<HIDD>(hact, HIDD, mt * 16 + l15, wT, HIDD, nq * 64, l15, hf, acc);
#pragma unroll
  for (int j = 0; j < 4; ++j) {
    int col = nq * 64 + j * 16 + l15;
    float bb = bias[col];
#pragma unroll
    for (int i = 0; i < 8; ++i) {
      int m = mt * 16 + hf * 8 + i;
      t[(size_t)m * CDIM + col] += acc[j][i] + bb;
    }
  }
}

// ---- final LN + NCHW scatter ---------------------------------------------
__global__ __launch_bounds__(128)
void lnf_out_kernel(const float* __restrict__ t, const float* __restrict__ g,
                    const float* __restrict__ bt, float* __restrict__ out) {
  __shared__ float red[128];
  size_t tok = blockIdx.x;
  int c = threadIdx.x;
  float v = t[tok * CDIM + c];
  red[c] = v; __syncthreads();
  for (int s = 64; s > 0; s >>= 1) { if (c < s) red[c] += red[c + s]; __syncthreads(); }
  float mu = red[0] * (1.0f / CDIM); __syncthreads();
  float d = v - mu;
  red[c] = d * d; __syncthreads();
  for (int s = 64; s > 0; s >>= 1) { if (c < s) red[c] += red[c + s]; __syncthreads(); }
  float var = red[0] * (1.0f / CDIM);
  float res = d * rsqrtf(var + 1e-6f) * g[c] + bt[c];
  size_t b = tok / NTOK;
  size_t hw = tok % NTOK;
  out[((b * CDIM) + c) * NTOK + hw] = res;
}

// ---------------------------------------------------------------------------
static inline size_t alignup(size_t v) { return (v + 255) & ~(size_t)255; }

extern "C" void kernel_launch(void* const* d_in, const int* in_sizes, int n_in,
                              void* d_out, int out_size, void* d_ws, size_t ws_size,
                              hipStream_t stream) {
  (void)in_sizes; (void)n_in; (void)out_size; (void)ws_size;
  const float* x      = (const float*)d_in[0];
  const float* pe_w   = (const float*)d_in[1];
  const float* pe_b   = (const float*)d_in[2];
  const float* pos_w  = (const float*)d_in[3];
  const float* pos_b  = (const float*)d_in[4];
  const float* peln_g = (const float*)d_in[5];
  const float* peln_b = (const float*)d_in[6];
  const float* ln1_g  = (const float*)d_in[7];
  const float* ln1_b  = (const float*)d_in[8];
  const float* q_w    = (const float*)d_in[9];
  const float* q_b    = (const float*)d_in[10];
  const float* kv_w   = (const float*)d_in[11];
  const float* kv_b   = (const float*)d_in[12];
  const float* ln2_g  = (const float*)d_in[13];
  const float* ln2_b  = (const float*)d_in[14];
  const float* fc1_w  = (const float*)d_in[15];
  const float* fc1_b  = (const float*)d_in[16];
  const float* dw_w   = (const float*)d_in[17];
  const float* dw_b   = (const float*)d_in[18];
  const float* fc2_w  = (const float*)d_in[19];
  const float* fc2_b  = (const float*)d_in[20];
  const float* lnf_g  = (const float*)d_in[21];
  const float* lnf_b  = (const float*)d_in[22];
  float* out = (float*)d_out;

  char* ws = (char*)d_ws;
  size_t off = 0;
  float*    t      = (float*)(ws + off);    off += alignup((size_t)NT * CDIM * 4);
  _Float16* h16    = (_Float16*)(ws + off); off += alignup((size_t)NT * CDIM * 2);
  float*    zbuf   = (float*)(ws + off);    off += alignup((size_t)NT * 4);
  float*    ksum   = (float*)(ws + off);    off += alignup((size_t)BB * CDIM * 4);
  _Float16* kvt16  = (_Float16*)(ws + off); off += alignup((size_t)BB * CDIM * CDIM * 2);
  _Float16* wqkvT  = (_Float16*)(ws + off); off += alignup((size_t)384 * CDIM * 2);
  _Float16* fc1wT  = (_Float16*)(ws + off); off += alignup((size_t)HIDD * CDIM * 2);
  _Float16* fc2wT  = (_Float16*)(ws + off); off += alignup((size_t)CDIM * HIDD * 2);
  size_t base2 = off;
  // phase-1 region: y, Q16, K16, V16 (dead after attn)
  float*    y      = (float*)(ws + base2);
  _Float16* Q16    = (_Float16*)(ws + base2 + alignup((size_t)BB * CDIM * NTOK * 4));
  _Float16* K16    = Q16 + (size_t)NT * CDIM;
  _Float16* V16    = K16 + (size_t)NT * CDIM;
  // phase-2 region overlays phase-1: hmid (f32), hact (f16)
  float*    hmid   = (float*)(ws + base2);
  _Float16* hact   = (_Float16*)(ws + base2 + alignup((size_t)NT * HIDD * 4));

  // weight transposes/conversions
  wtrans_kernel<<<(128 * 128 + 255) / 256, 256, 0, stream>>>(q_w,  wqkvT,              CDIM, CDIM);
  wtrans_kernel<<<(128 * 256 + 255) / 256, 256, 0, stream>>>(kv_w, wqkvT + 128 * CDIM, CDIM, 256);
  wtrans_kernel<<<(128 * 512 + 255) / 256, 256, 0, stream>>>(fc1_w, fc1wT,             CDIM, HIDD);
  wtrans_kernel<<<(512 * 128 + 255) / 256, 256, 0, stream>>>(fc2_w, fc2wT,             HIDD, CDIM);

  patch_embed_kernel<<<BB * HO, 256, 0, stream>>>(x, pe_w, pe_b, y);

  {
    size_t tot = (size_t)BB * CDIM * NTOK;
    posgate_kernel<<<(unsigned)((tot + 255) / 256), 256, 0, stream>>>(y, pos_w, pos_b, t);
  }

  tokln_kernel<<<NT, 128, 0, stream>>>(t, peln_g, peln_b, ln1_g, ln1_b, h16);

  qkv_kernel<<<(NT / 16) * 6 / 8, 256, 0, stream>>>(h16, wqkvT, q_b, kv_b, Q16, K16, V16);

  ksum_kernel<<<BB, 128, 0, stream>>>(K16, ksum);
  kv_outer_kernel<<<BB, 256, 0, stream>>>(K16, V16, kvt16);
  zcalc_kernel<<<NT / 256, 256, 0, stream>>>(Q16, ksum, zbuf);
  attn_kernel<<<(NT / 16) * 2 / 8, 256, 0, stream>>>(Q16, kvt16, zbuf, t);

  ln_kernel<<<NT, 128, 0, stream>>>(t, ln2_g, ln2_b, h16, 1e-6f);

  fc1_kernel<<<(NT / 16) * 8 / 8, 256, 0, stream>>>(h16, fc1wT, fc1_b, hmid);

  {
    size_t tot = (size_t)NT * HIDD;
    dwgelu_kernel<<<(unsigned)((tot + 255) / 256), 256, 0, stream>>>(hmid, dw_w, dw_b, hact);
  }

  fc2_kernel<<<(NT / 16) * 2 / 8, 256, 0, stream>>>(hact, fc2wT, fc2_b, t);

  lnf_out_kernel<<<NT, 128, 0, stream>>>(t, lnf_g, lnf_b, out);
}
